// QwenAttention_64665027609326
// MI455X (gfx1250) — matrix operand order repaired and checked by reading, not compiled, not run on hardware
//
#include <hip/hip_runtime.h>

typedef __bf16 bf16_t;
typedef __attribute__((ext_vector_type(16))) __bf16 v16bf;
typedef __attribute__((ext_vector_type(8)))  __bf16 v8bf;
typedef __attribute__((ext_vector_type(8)))  float  v8f;

__device__ __forceinline__ v16bf cat8(v8bf lo, v8bf hi) {
  return __builtin_shufflevector(lo, hi, 0,1,2,3,4,5,6,7,8,9,10,11,12,13,14,15);
}

__device__ __forceinline__ v8f wmma_bf16(v16bf a, v16bf b, v8f c) {
  return __builtin_amdgcn_wmma_f32_16x16x32_bf16(
      /*neg_a=*/false, a, /*neg_b=*/false, b,
      /*c_mod=*/(short)0, c, /*reuse_a=*/false, /*reuse_b=*/false);
}

// ---------------------------------------------------------------------------
// Binary-quantized weight dequant: w = sign(latent) * (softplus(raw)+1e-6)
// latent: [out_dim, in_dim] f32, raws: [out_dim, in_dim/128] f32 -> bf16
// ---------------------------------------------------------------------------
__global__ __launch_bounds__(256) void quant_kernel(const float* __restrict__ latent,
                                                    const float* __restrict__ raws,
                                                    bf16_t* __restrict__ out,
                                                    int out_dim, int in_dim) {
  int idx = blockIdx.x * 256 + threadIdx.x;
  if (idx >= out_dim * in_dim) return;
  int o = idx / in_dim;
  int i = idx - o * in_dim;
  float r = raws[o * (in_dim >> 7) + (i >> 7)];
  float sp = (r > 20.f) ? r : log1pf(expf(r));
  float s = sp + 1e-6f;
  out[idx] = (bf16_t)((latent[idx] >= 0.f) ? s : -s);
}

__global__ __launch_bounds__(256) void conv_kernel(const float* __restrict__ in,
                                                   bf16_t* __restrict__ out, int n) {
  int idx = blockIdx.x * 256 + threadIdx.x;
  if (idx < n) out[idx] = (bf16_t)in[idx];
}

// ---------------------------------------------------------------------------
// C[M,N] = A[M,K] (bf16, row-major) * W[N,K]^T (bf16, row-major)
// Block: 256 threads = 8 waves, block tile 128(M) x 64(N); wave tile 16x64.
// K-step 32 -> v_wmma_f32_16x16x32_bf16.
// ---------------------------------------------------------------------------
__global__ __launch_bounds__(256) void gemm_xwT(const bf16_t* __restrict__ A,
                                                const bf16_t* __restrict__ W,
                                                void* __restrict__ Cout,
                                                int M, int N, int K, int f32out) {
  int tid = threadIdx.x;
  int w = tid >> 5, lane = tid & 31;
  int lmod = lane & 15, lhi = lane >> 4;
  int row0 = blockIdx.x * 128 + w * 16;
  int col0 = blockIdx.y * 64;

  v8f acc[4];
  {
    v8f z = {0.f,0.f,0.f,0.f,0.f,0.f,0.f,0.f};
    acc[0] = z; acc[1] = z; acc[2] = z; acc[3] = z;
  }

  const bf16_t* Arow = A + (size_t)(row0 + lmod) * K;
  for (int k0 = 0; k0 < K; k0 += 32) {
    // Unconditional next-chunk prefetch (global_prefetch_b8). Speculative
    // prefetch past the buffer end is silently dropped (ISA 10.5), so no
    // branch is needed and the K-loop stays a straight line for pipelining.
    __builtin_prefetch(Arow + k0 + 128, 0, 1);
    // A fragment (16x32 bf16): lane = row (M), lanes>=16 take K offset +16; 16 contiguous K
    int ka = k0 + lhi * 16;
    v8bf alo = *(const v8bf*)(Arow + ka);
    v8bf ahi = *(const v8bf*)(Arow + ka + 8);
    v16bf af = cat8(alo, ahi);
    int kb = k0 + lhi * 16;
#pragma unroll
    for (int j = 0; j < 4; ++j) {
      // B fragment (32x16 bf16): lane%16 = N col = W row; 16 contiguous K
      const bf16_t* wp = W + (size_t)(col0 + j * 16 + lmod) * K + kb;
      v16bf bfv = *(const v16bf*)wp;
      acc[j] = wmma_bf16(af, bfv, acc[j]);
    }
  }

#pragma unroll
  for (int j = 0; j < 4; ++j) {
#pragma unroll
    for (int v = 0; v < 8; ++v) {
      int r = row0 + v + 8 * lhi;
      int c = col0 + j * 16 + lmod;
      if (f32out) ((float*)Cout)[(size_t)r * N + c] = acc[j][v];
      else        ((bf16_t*)Cout)[(size_t)r * N + c] = (bf16_t)acc[j][v];
    }
  }
}

// ---------------------------------------------------------------------------
// RoPE (first 16 dims of each head, base 1e6) + split/transpose of fused QKV
// qkv: [4096, 1024] bf16 -> Q[B,H,S,64], K[B,KV,S,64] (roped), V[B,KV,S,64]
// ---------------------------------------------------------------------------
__global__ __launch_bounds__(256) void rope_split_kernel(const bf16_t* __restrict__ qkv,
                                                         bf16_t* __restrict__ Qb,
                                                         bf16_t* __restrict__ Kb,
                                                         bf16_t* __restrict__ Vb) {
  int idx = blockIdx.x * 256 + threadIdx.x;
  if (idx >= 4096 * 1024) return;
  int m = idx >> 10, n = idx & 1023;
  int b = m >> 11, s = m & 2047;
  const bf16_t* row = qkv + (size_t)m * 1024;
  int d = n & 63;
  float val;
  if (n < 896 && d < 16) {  // q or k head, roped dims
    int base = n - d;
    int i = d >> 1;
    float inv = powf(1.0e6f, -((float)(2 * i)) / 16.f);
    float fr = (float)s * inv;
    float c = cosf(fr), sn = sinf(fr);
    float x1 = (float)row[base + 2 * i];
    float x2 = (float)row[base + 2 * i + 1];
    val = (d & 1) ? (x1 * sn + x2 * c) : (x1 * c - x2 * sn);
  } else {
    val = (float)row[n];
  }
  bf16_t ov = (bf16_t)val;
  if (n < 768) {
    int h = n >> 6;
    Qb[(((size_t)b * 12 + h) * 2048 + s) * 64 + d] = ov;
  } else if (n < 896) {
    int kvh = (n - 768) >> 6;
    Kb[(((size_t)b * 2 + kvh) * 2048 + s) * 64 + d] = ov;
  } else {
    int kvh = (n - 896) >> 6;
    Vb[(((size_t)b * 2 + kvh) * 2048 + s) * 64 + d] = ov;
  }
}

// ---------------------------------------------------------------------------
// Flash attention, causal, GQA (6 q-heads per kv-head), scale = 1/8.
// Grid: B*H*(S/128). Block: 256 thr = 8 waves; wave owns 16 query rows.
// K tile (64x64) streamed memory->LDS via GLOBAL_LOAD_ASYNC_TO_LDS_B128
// (ASYNCcnt, no VGPR round trip). V tile transposed in LDS (Vt[d][k]) so
// PV B-frags are contiguous; P goes through per-wave LDS for C->A layout.
// ---------------------------------------------------------------------------
__global__ __launch_bounds__(256) void attn_kernel(const bf16_t* __restrict__ Q,
                                                   const bf16_t* __restrict__ K,
                                                   const bf16_t* __restrict__ V,
                                                   bf16_t* __restrict__ O) {
  __shared__ bf16_t Kt[64 * 64];        // [k][d]
  __shared__ bf16_t Vt[64 * 64];        // [d][k] (transposed)
  __shared__ bf16_t Pt[8][16 * 64];     // per-wave P scratch [row][k]

  int blk = blockIdx.x;
  int qb = blk & 15;
  int bh = blk >> 4;
  int h = bh % 12, b = bh / 12;
  int kvh = h / 6;
  int tid = threadIdx.x, w = tid >> 5, lane = tid & 31;
  int lmod = lane & 15, lhi = lane >> 4;
  int q0 = qb * 128 + w * 16;

  const bf16_t* Qbase = Q + ((size_t)b * 12 + h) * 2048 * 64;
  const bf16_t* Kbase = K + ((size_t)b * 2 + kvh) * 2048 * 64;
  const bf16_t* Vbase = V + ((size_t)b * 2 + kvh) * 2048 * 64;

  v8f of[4];
  {
    v8f z = {0.f,0.f,0.f,0.f,0.f,0.f,0.f,0.f};
    of[0] = z; of[1] = z; of[2] = z; of[3] = z;
  }
  float rowm[8], rowl[8];
#pragma unroll
  for (int v = 0; v < 8; ++v) { rowm[v] = -3.0e38f; rowl[v] = 0.f; }

  // Preload this wave's Q fragments (16 rows x 64 d = two 16x32 A-frags)
  v16bf qf[2];
#pragma unroll
  for (int ds2 = 0; ds2 < 2; ++ds2) {
    const bf16_t* qp = Qbase + (size_t)(q0 + lmod) * 64 + ds2 * 32 + lhi * 16;
    v8bf lo = *(const v8bf*)qp;
    v8bf hi = *(const v8bf*)(qp + 8);
    qf[ds2] = cat8(lo, hi);
  }

  int nk = qb * 2 + 2;  // causal: k tiles up to and including diagonal block
  for (int kt = 0; kt < nk; ++kt) {
    __syncthreads();
    {  // K tile: async DMA memory -> LDS (32 B per thread, two b128 issues;
       // IOFFSET applies to both global and LDS sides).
      int r = tid >> 2;
      int c0 = (tid & 3) * 16;
      const bf16_t* kp = Kbase + (size_t)(kt * 64 + r) * 64 + c0;
      unsigned ldsoff = (unsigned)(uintptr_t)(&Kt[r * 64 + c0]);
      unsigned long long ga = (unsigned long long)(uintptr_t)kp;
      asm volatile("global_load_async_to_lds_b128 %0, %1, off"
                   :: "v"(ldsoff), "v"(ga) : "memory");
      asm volatile("global_load_async_to_lds_b128 %0, %1, off offset:16"
                   :: "v"(ldsoff), "v"(ga) : "memory");
      // V tile: load to VGPRs and transpose while staging into LDS
      const bf16_t* vp = Vbase + (size_t)(kt * 64 + r) * 64 + c0;
      v16bf vd = *(const v16bf*)vp;
#pragma unroll
      for (int e = 0; e < 16; ++e) Vt[(c0 + e) * 64 + r] = vd[e];
      // drain this wave's async transfers; barrier below covers other waves
      asm volatile("s_wait_asynccnt 0x0" ::: "memory");
    }
    __syncthreads();

    // scores = Q x K^T : 16 rows x 64 k-cols
    v8f sf[4];
    {
      v8f z = {0.f,0.f,0.f,0.f,0.f,0.f,0.f,0.f};
      sf[0] = z; sf[1] = z; sf[2] = z; sf[3] = z;
    }
#pragma unroll
    for (int ds2 = 0; ds2 < 2; ++ds2) {
#pragma unroll
      for (int j = 0; j < 4; ++j) {
        const bf16_t* bp = &Kt[(j * 16 + lmod) * 64 + ds2 * 32 + lhi * 16];
        v16bf bfv = *(const v16bf*)bp;
        sf[j] = wmma_bf16(qf[ds2], bfv, sf[j]);
      }
    }

    // online softmax (row = v + 8*lhi within this wave's 16 rows)
#pragma unroll
    for (int v = 0; v < 8; ++v) {
      int qi = q0 + v + 8 * lhi;
      float sv[4];
      float mx = -3.0e38f;
#pragma unroll
      for (int j = 0; j < 4; ++j) {
        int kj = kt * 64 + j * 16 + lmod;
        float sc = sf[j][v] * 0.125f;
        if (kj > qi) sc = -1.0e30f;  // causal mask (matches NEG in reference)
        sv[j] = sc;
        mx = fmaxf(mx, sc);
      }
#pragma unroll
      for (int off = 1; off < 16; off <<= 1) mx = fmaxf(mx, __shfl_xor(mx, off, 32));
      float mnew = fmaxf(rowm[v], mx);
      float corr = expf(rowm[v] - mnew);
      float rs = 0.f;
#pragma unroll
      for (int j = 0; j < 4; ++j) {
        float p = expf(sv[j] - mnew);
        rs += p;
        Pt[w][(v + 8 * lhi) * 64 + j * 16 + lmod] = (bf16_t)p;
      }
#pragma unroll
      for (int off = 1; off < 16; off <<= 1) rs += __shfl_xor(rs, off, 32);
      rowl[v] = rowl[v] * corr + rs;
      rowm[v] = mnew;
#pragma unroll
      for (int jd = 0; jd < 4; ++jd) of[jd][v] = of[jd][v] * corr;
    }
    __syncthreads();  // make Pt visible for A-fragment reload

    // O += P x V : P is 16x64(k), V^T staged as Vt[d][k]
#pragma unroll
    for (int ks = 0; ks < 2; ++ks) {
      const bf16_t* pr = &Pt[w][lmod * 64 + ks * 32 + lhi * 16];
      v8bf lo = *(const v8bf*)pr;
      v8bf hi = *(const v8bf*)(pr + 8);
      v16bf pf = cat8(lo, hi);
#pragma unroll
      for (int jd = 0; jd < 4; ++jd) {
        const bf16_t* vv = &Vt[(jd * 16 + lmod) * 64 + ks * 32 + lhi * 16];
        v16bf vfv = *(const v16bf*)vv;
        of[jd] = wmma_bf16(pf, vfv, of[jd]);
      }
    }
  }

  // epilogue: normalize and write [B*S, 768] bf16 for the O-projection GEMM
#pragma unroll
  for (int v = 0; v < 8; ++v) {
    float inv = 1.f / rowl[v];
    int r = q0 + v + 8 * lhi;
#pragma unroll
    for (int jd = 0; jd < 4; ++jd) {
      O[((size_t)b * 2048 + r) * 768 + h * 64 + jd * 16 + lmod] =
          (bf16_t)(of[jd][v] * inv);
    }
  }
}

// ---------------------------------------------------------------------------
extern "C" void kernel_launch(void* const* d_in, const int* in_sizes, int n_in,
                              void* d_out, int out_size, void* d_ws, size_t ws_size,
                              hipStream_t stream) {
  (void)in_sizes; (void)n_in; (void)out_size; (void)ws_size;
  const float* x      = (const float*)d_in[0];
  const float* q_lat  = (const float*)d_in[1];
  const float* q_raw  = (const float*)d_in[2];
  const float* k_lat  = (const float*)d_in[3];
  const float* k_raw  = (const float*)d_in[4];
  const float* v_lat  = (const float*)d_in[5];
  const float* v_raw  = (const float*)d_in[6];
  const float* o_lat  = (const float*)d_in[7];
  const float* o_raw  = (const float*)d_in[8];
  float* out = (float*)d_out;

  char* ws = (char*)d_ws;
  size_t off = 0;
  auto alloc = [&](size_t bytes) -> void* {
    void* p = ws + off;
    off += (bytes + 255) & ~(size_t)255;
    return p;
  };

  bf16_t* Wqkv = (bf16_t*)alloc((size_t)1024 * 768 * 2);  // rows: 768 Q, 128 K, 128 V
  bf16_t* Wo   = (bf16_t*)alloc((size_t)768 * 768 * 2);
  bf16_t* xb   = (bf16_t*)alloc((size_t)4096 * 768 * 2);
  bf16_t* qkv  = (bf16_t*)alloc((size_t)4096 * 1024 * 2);
  bf16_t* Qb   = (bf16_t*)alloc((size_t)2 * 12 * 2048 * 64 * 2);
  bf16_t* Kb   = (bf16_t*)alloc((size_t)2 * 2 * 2048 * 64 * 2);
  bf16_t* Vb   = (bf16_t*)alloc((size_t)2 * 2 * 2048 * 64 * 2);
  bf16_t* Ab   = (bf16_t*)alloc((size_t)4096 * 768 * 2);

  // 1. dequantize weights to bf16 (QKV fused)
  quant_kernel<<<(768 * 768 + 255) / 256, 256, 0, stream>>>(q_lat, q_raw, Wqkv, 768, 768);
  quant_kernel<<<(128 * 768 + 255) / 256, 256, 0, stream>>>(k_lat, k_raw, Wqkv + (size_t)768 * 768, 128, 768);
  quant_kernel<<<(128 * 768 + 255) / 256, 256, 0, stream>>>(v_lat, v_raw, Wqkv + (size_t)896 * 768, 128, 768);
  quant_kernel<<<(768 * 768 + 255) / 256, 256, 0, stream>>>(o_lat, o_raw, Wo, 768, 768);

  // 2. x -> bf16
  conv_kernel<<<(4096 * 768 + 255) / 256, 256, 0, stream>>>(x, xb, 4096 * 768);

  // 3. fused QKV projection: [4096,768] x [1024,768]^T -> [4096,1024]
  {
    dim3 g(32, 16);
    gemm_xwT<<<g, 256, 0, stream>>>(xb, Wqkv, qkv, 4096, 1024, 768, 0);
  }

  // 4. RoPE + head split/transpose
  rope_split_kernel<<<(4096 * 1024) / 256, 256, 0, stream>>>(qkv, Qb, Kb, Vb);

  // 5. causal flash attention
  attn_kernel<<<2 * 12 * 16, 256, 0, stream>>>(Qb, Kb, Vb, Ab);

  // 6. output projection -> f32: [4096,768] x [768,768]^T
  {
    dim3 g(32, 12);
    gemm_xwT<<<g, 256, 0, stream>>>(Ab, Wo, out, 4096, 768, 768, 1);
  }
}
